// SRKNN_58823872086380
// MI455X (gfx1250) — compile-verified, matrix operand order checked by reading, not run on hardware
//
#include <hip/hip_runtime.h>
#include <hip/hip_bf16.h>
#include <math.h>

// ---------------------------------------------------------------------------
// SRKNN on MI455X (gfx1250, wave32).
//
// dis^2(i,j) = s^T A s,  s = sigmoid(pm_i - ps_j) in R^64,  A = M^T M.
// Per 16-pair tile (one i-row x 16 j's):
//   H = A_metric * S^T        (4x  16x16 tiles, 8 WMMA f16 K=32)
//   G = S * H = S A S^T       (1   16x16 tile,  2 WMMA accumulating)
//   diag(G) = the 16 squared distances.
// The S fragment is reused as both A-operand and B-operand (A-frag and B-frag
// per-lane layouts coincide for a transposed matrix), and H's C-layout output
// repacks in-lane into the B-fragment of the second GEMM. No shuffles.
//
// VALU budget: 32 sigmoids/lane/tile dominate; use HW v_rcp_f32/v_sqrt_f32
// instead of IEEE division (results are truncated to f16 anyway) so they can
// co-execute with the WMMA (XDL-as-TRANS) stream.
// ---------------------------------------------------------------------------

typedef __attribute__((ext_vector_type(16))) _Float16 v16h;
typedef __attribute__((ext_vector_type(8)))  float    v8f;

#define BN    2
#define NPTS  1024
#define C_IN  128
#define C2    64
#define TOPK  16

// K-index pattern inside a 16-bit WMMA 16x32 A-fragment / 32x16 B-fragment
// (CDNA5 ISA 7.12.2): vgpr vb in 0..7 holds 2 packed halves, lanes split K.
__device__ __forceinline__ int frag_k(int kc, int vb, int half) {
  return ((vb < 4) ? 0 : 16) + 8 * half + 2 * (vb & 3) + 32 * kc;
}

// sigmoid(x) = rcp(1 + exp(-x)) using v_exp_f32 + v_rcp_f32 (no IEEE div)
__device__ __forceinline__ float fast_sigmoid(float x) {
  return __builtin_amdgcn_rcpf(1.f + __expf(-x));
}

// --------------------------- A = M^T M (f16) -------------------------------
__global__ void srknn_metric(const float* __restrict__ M,
                             _Float16* __restrict__ A16) {
  int idx = blockIdx.x * blockDim.x + threadIdx.x;   // 0..4095
  int d = idx >> 6, e = idx & 63;
  float acc = 0.f;
  #pragma unroll 4
  for (int k = 0; k < C2; ++k) acc += M[k * C2 + d] * M[k * C2 + e];
  A16[idx] = (_Float16)acc;
}

// ---------------- projections + sign-preserving log map --------------------
// grid (n=1024, b=2, which=2), block 64 (o). X is [B,C,H,W] -> XV[b,n,c].
__global__ void srknn_proj(const float* __restrict__ X,
                           const float* __restrict__ Wm,
                           const float* __restrict__ Ws,
                           float* __restrict__ pm, float* __restrict__ ps) {
  __shared__ float xrow[C_IN];
  const int n = blockIdx.x, b = blockIdx.y, which = blockIdx.z;
  const int o = threadIdx.x;                          // 0..63
  xrow[o]      = X[(size_t)b * C_IN * NPTS + (size_t)o        * NPTS + n];
  xrow[o + 64] = X[(size_t)b * C_IN * NPTS + (size_t)(o + 64) * NPTS + n];
  __syncthreads();
  const float* W = which ? Ws : Wm;
  float acc = 0.f;
  #pragma unroll 4
  for (int c = 0; c < C_IN; ++c) acc += xrow[c] * W[o * C_IN + c];
  float sgn = (acc > 0.f) ? 1.f : ((acc < 0.f) ? -1.f : 0.f);
  float y = sgn * __logf(fabsf(acc) + 1e-10f);
  float* dst = which ? ps : pm;
  dst[((size_t)b * NPTS + n) * C2 + o] = y;
}

// ------------------------------ main kernel --------------------------------
// grid (iTile=64, b=2), block 128 (4 waves). Wave w owns rows w, w+4, w+8, w+12.
__global__ void __launch_bounds__(128)
srknn_main(const float* __restrict__ pm, const float* __restrict__ ps,
           const _Float16* __restrict__ A16,
           int* __restrict__ outIdx, float* __restrict__ outVal) {
  __shared__ float pmT[16 * C2];          // i-tile of pm (4 KB)
  __shared__ float Gt[4][16][16];         // per-wave G tile   (4 KB)
  __shared__ float rowbuf[4][NPTS];       // per-wave diff row (16 KB)

  const int iTile = blockIdx.x;
  const int b     = blockIdx.y;
  const int tid   = threadIdx.x;
  const int lane  = tid & 31;
  const int wv    = tid >> 5;
  const int half  = lane >> 4;
  const int m16   = lane & 15;

  {   // cooperative, coalesced load of the 16x64 pm tile
    const float* src = pm + ((size_t)b * NPTS + iTile * 16) * C2;
    for (int t = tid; t < 16 * C2; t += 128) pmT[t] = src[t];
  }
  __syncthreads();

  // Preload metric A-fragments (constant for whole kernel): 64 VGPRs.
  // metF[mt][ec]: rows d = mt*16 + m16, K = e per fragment pattern.
  v16h metF[4][2];
  #pragma unroll
  for (int mt = 0; mt < 4; ++mt)
    #pragma unroll
    for (int ec = 0; ec < 2; ++ec)
      #pragma unroll
      for (int vb = 0; vb < 8; ++vb) {
        int e = frag_k(ec, vb, half);
        const _Float16* ap = A16 + (mt * 16 + m16) * C2 + e;
        metF[mt][ec][2 * vb]     = ap[0];
        metF[mt][ec][2 * vb + 1] = ap[1];
      }

  const v8f vzero = {};

  for (int rr = 0; rr < 4; ++rr) {
    const int rg = wv + 4 * rr;             // local i-row 0..15
    const int ig = iTile * 16 + rg;         // global i

    // pm_i values in fragment order; reused across all 64 j-tiles.
    float pmv[2][16];
    #pragma unroll
    for (int kc = 0; kc < 2; ++kc)
      #pragma unroll
      for (int vb = 0; vb < 8; ++vb) {
        int k = frag_k(kc, vb, half);
        pmv[kc][2 * vb]     = pmT[rg * C2 + k];
        pmv[kc][2 * vb + 1] = pmT[rg * C2 + k + 1];
      }

    for (int jb = 0; jb < NPTS / 16; ++jb) {
      const int j = jb * 16 + m16;
      const float* psr = ps + ((size_t)b * NPTS + j) * C2;

      // S fragment: S[p][k] = sigmoid(pm_i[k] - ps_j[k]), pair p = m16.
      // exp + rcp co-execute with the matrix pipe.
      v16h sF[2];
      #pragma unroll
      for (int kc = 0; kc < 2; ++kc)
        #pragma unroll
        for (int vb = 0; vb < 8; ++vb) {
          int k = frag_k(kc, vb, half);
          float2 pv = *(const float2*)(psr + k);       // 8B aligned (k even)
          float s0 = fast_sigmoid(pmv[kc][2 * vb]     - pv.x);
          float s1 = fast_sigmoid(pmv[kc][2 * vb + 1] - pv.y);
          sF[kc][2 * vb]     = (_Float16)s0;
          sF[kc][2 * vb + 1] = (_Float16)s1;
        }

      // H = A_metric * S^T : contraction over e, sF used as B-operand.
      v8f hh[4];
      #pragma unroll
      for (int mt = 0; mt < 4; ++mt) {
        hh[mt] = __builtin_amdgcn_wmma_f32_16x16x32_f16(
            false, metF[mt][0], false, sF[0], (short)0, vzero, false, false);
        hh[mt] = __builtin_amdgcn_wmma_f32_16x16x32_f16(
            false, metF[mt][1], false, sF[1], (short)0, hh[mt], false, false);
      }

      // In-lane repack: H C-layout -> B-fragments of G (d-chunks of 32).
      v16h bg0, bg1;
      #pragma unroll
      for (int i = 0; i < 8; ++i) {
        bg0[i]     = (_Float16)hh[0][i];
        bg0[8 + i] = (_Float16)hh[1][i];
        bg1[i]     = (_Float16)hh[2][i];
        bg1[8 + i] = (_Float16)hh[3][i];
      }

      // G = S * H = S A S^T ; only the diagonal is needed.
      v8f g = __builtin_amdgcn_wmma_f32_16x16x32_f16(
          false, sF[0], false, bg0, (short)0, vzero, false, false);
      g = __builtin_amdgcn_wmma_f32_16x16x32_f16(
          false, sF[1], false, bg1, (short)0, g, false, false);

      // diag extraction through per-wave LDS tile (in-order DS per wave).
      #pragma unroll
      for (int v = 0; v < 8; ++v) Gt[wv][v + 8 * half][m16] = g[v];
      asm volatile("s_wait_dscnt 0x0" ::: "memory");
      if (lane < 16) {
        float d2  = Gt[wv][lane][lane];
        float dis = __builtin_amdgcn_sqrtf(fmaxf(d2, 0.f));
        rowbuf[wv][jb * 16 + lane] =
            __builtin_amdgcn_rcpf(1.f + __expf(dis));   // sigmoid(-dis)
      }
      asm volatile("s_wait_dscnt 0x0" ::: "memory");
    }

    // Stable descending top-16 over this row (ties -> lowest j, like top_k).
    if (lane == 0) {
      float tv[TOPK]; int ti[TOPK];
      #pragma unroll
      for (int k = 0; k < TOPK; ++k) { tv[k] = -3.4e38f; ti[k] = 0; }
      for (int j2 = 0; j2 < NPTS; ++j2) {
        float v = rowbuf[wv][j2];
        if (v > tv[TOPK - 1]) {
          int p = TOPK - 1;
          while (p > 0 && v > tv[p - 1]) {
            tv[p] = tv[p - 1]; ti[p] = ti[p - 1]; --p;
          }
          tv[p] = v; ti[p] = j2;
        }
      }
      size_t base = ((size_t)b * NPTS + ig) * TOPK;
      for (int k = 0; k < TOPK; ++k) {
        outIdx[base + k] = ti[k];
        outVal[base + k] = -tv[k];        // reference negates values
      }
    }
  }
}

// ---------------------------------------------------------------------------
extern "C" void kernel_launch(void* const* d_in, const int* in_sizes, int n_in,
                              void* d_out, int out_size, void* d_ws, size_t ws_size,
                              hipStream_t stream) {
  const float* X    = (const float*)d_in[0];   // [2,128,32,32]
  const float* Wmin = (const float*)d_in[1];   // [64,128]
  const float* Wsub = (const float*)d_in[2];   // [64,128]
  const float* M    = (const float*)d_in[3];   // [64,64]

  // workspace: pm (512KB) | ps (512KB) | A16 (8KB)
  float*     pm  = (float*)d_ws;
  float*     ps  = pm + (size_t)BN * NPTS * C2;
  _Float16*  A16 = (_Float16*)(ps + (size_t)BN * NPTS * C2);

  // output: index [2,1024*16] int32, then value [2,1024*16,1] f32 (flat concat)
  int*   outIdx = (int*)d_out;
  float* outVal = (float*)d_out + (size_t)BN * NPTS * TOPK;

  srknn_metric<<<dim3(16), dim3(256), 0, stream>>>(M, A16);
  srknn_proj<<<dim3(NPTS, BN, 2), dim3(64), 0, stream>>>(X, Wmin, Wsub, pm, ps);
  srknn_main<<<dim3(NPTS / 16, BN), dim3(128), 0, stream>>>(pm, ps, A16,
                                                            outIdx, outVal);
}
